// VoxelMlp_45062796869781
// MI455X (gfx1250) — matrix operand order repaired and checked by reading, compiled, not run on hardware
//
#include <hip/hip_runtime.h>
#include <hip/hip_bf16.h>
#include <math.h>

// ---------------------------------------------------------------- constants
#define P_TOT    524288
#define N_RAYS   4096
#define ACT_SHIFT (-2.1972245773362196f)   // log(1/(1-0.1)-1)
#define INTERVAL 0.5f

typedef _Float16 half_t;
typedef __attribute__((ext_vector_type(16))) _Float16 v16h;
typedef __attribute__((ext_vector_type(8)))  float    v8f;

// Weight-fragment offsets in halfs inside d_ws (each frag = 512 halfs)
#define OFF_WB0    0        // 33->64  (Kpad 64)  8 frags
#define OFF_WB1    4096     // 64->64             8
#define OFF_WB2    8192     // 64->64             8
#define OFF_WA0T   12288    // 33->64  (Kpad 64)  8   rows 0..32 of wa0
#define OFF_WA0B   16384    // 64->64             8   rows 33..96 of wa0
#define OFF_WA1    20480    // 64->64             8
#define OFF_WA2    24576    // 64->64             8
#define OFF_WLAT   28672    // 64->64             8
#define OFF_WVT    32768    // 64->32             4   rows 0..63 of w_view
#define OFF_WVB    34816    // 21->32  (Kpad 32)  2   rows 64..84 of w_view
#define OFF_WC0    35840    // 32->16             1
#define WF_TOTAL_HALFS 36352
#define RGB_WS_OFF 73728    // byte offset of per-point rgb (P*3 f32)

// ============================================================== prep kernel
// Repack f32 weights into f16 B-fragments for v_wmma_f32_16x16x32_f16:
//   frag(kc,nt): lane holds column n = nt*16+(lane&15);
//   halfs i=0..15 hold K = kc*32 + (lane>=16 ? 16 : 0) + i.  Zero pad K>=K_real.
__global__ void prep_weights(const float* wb0, const float* wb1, const float* wb2,
                             const float* wa0, const float* wa1, const float* wa2,
                             const float* wlat, const float* wview, const float* wc0,
                             half_t* dst) {
    const int FSTART[12] = {0, 8, 16, 24, 32, 40, 48, 56, 64, 68, 70, 71};
    const int KR[11]   = {33, 64, 64, 33, 64, 64, 64, 64, 64, 21, 32};
    const int NNT[11]  = {4, 4, 4, 4, 4, 4, 4, 4, 2, 2, 1};
    const int ROFF[11] = {0, 0, 0, 0, 33, 0, 0, 0, 0, 64, 0};
    const int SRCN[11] = {64, 64, 64, 64, 64, 64, 64, 64, 32, 32, 16};
    const int SRCI[11] = {0, 1, 2, 3, 3, 4, 5, 6, 7, 7, 8};
    const float* srcs[9] = {wb0, wb1, wb2, wa0, wa1, wa2, wlat, wview, wc0};

    int h = blockIdx.x * blockDim.x + threadIdx.x;
    if (h >= WF_TOTAL_HALFS) return;
    int f = h >> 9;                 // global fragment index
    int l = 0;
    while (f >= FSTART[l + 1]) ++l; // find layer
    int rel    = h - (FSTART[l] << 9);
    int fragIx = rel >> 9;
    int within = rel & 511;
    int lane   = within >> 4;
    int i      = within & 15;
    int nnt    = NNT[l];
    int kc     = fragIx / nnt;
    int nt     = fragIx - kc * nnt;
    int n      = (nt << 4) + (lane & 15);
    int k      = (kc << 5) + (((lane >> 4) & 1) << 4) + i;
    float v = 0.0f;
    if (k < KR[l]) v = srcs[SRCI[l]][(k + ROFF[l]) * SRCN[l] + n];
    dst[h] = (half_t)v;
}

// ============================================================== MLP kernel
// One wave owns a 32-point tile (two 16-row M tiles sharing every B fragment).
struct WaveLds {
    half_t E[32 * 64];   // coord embedding, K-padded to 64
    half_t H1[32 * 64];
    half_t H2[32 * 64];
    half_t V[32 * 32];   // view embedding, K-padded to 32
};

__device__ __forceinline__ v16h load_a_frag(const half_t* act, int pitchH, int kc, int lane) {
    // A 16x32 k order used here (matches B frags): lane&15 = row; VGPR r, half h ->
    //   K = (lane>=16?16:0) + r*2 + h   (pairs contiguous -> u32 loads)
    union { v16h v; unsigned u[8]; } r;
    int row = lane & 15;
    int hi  = (lane >> 4) & 1;
    const unsigned* rw = (const unsigned*)(act + row * pitchH);
#pragma unroll
    for (int g = 0; g < 8; ++g)
        r.u[g] = rw[kc * 16 + (hi << 3) + g];
    return r.v;
}

__device__ __forceinline__ v16h load_b_frag(const half_t* wf, int lane) {
    union { v16h v; uint4 q[2]; } r;
    const uint4* g = (const uint4*)(wf + lane * 16);
    r.q[0] = g[0];
    r.q[1] = g[1];
    return r.v;
}

// acc layout: acc[nt] = M-rows 0..15, acc[NNT+nt] = M-rows 16..31
template <int NKC, int NNT>
__device__ __forceinline__ void gemm_acc2(v8f* acc, const half_t* act, int pitchH,
                                          const half_t* wf, int lane) {
#pragma unroll
    for (int kc = 0; kc < NKC; ++kc) {
        v16h a0 = load_a_frag(act, pitchH, kc, lane);
        v16h a1 = load_a_frag(act + 16 * pitchH, pitchH, kc, lane);
#pragma unroll
        for (int nt = 0; nt < NNT; ++nt) {
            v16h b = load_b_frag(wf + (kc * NNT + nt) * 512, lane);  // one B, two WMMAs
            acc[nt] = __builtin_amdgcn_wmma_f32_16x16x32_f16(
                false, a0, false, b, (short)0, acc[nt], false, false);
            acc[NNT + nt] = __builtin_amdgcn_wmma_f32_16x16x32_f16(
                false, a1, false, b, (short)0, acc[NNT + nt], false, false);
        }
    }
}

template <int NNT, bool RELU>
__device__ __forceinline__ void store_act2(const v8f* acc, half_t* dst, int pitchH,
                                           const float* bias, int lane) {
    // C/D layout: VGPR r -> M = (lane>=16?8:0)+r, N = lane&15 (per 16-wide n-tile)
    int n0 = lane & 15;
    int mb = ((lane >> 4) & 1) ? 8 : 0;
#pragma unroll
    for (int mt = 0; mt < 2; ++mt) {
#pragma unroll
        for (int nt = 0; nt < NNT; ++nt) {
            float b = bias[nt * 16 + n0];
#pragma unroll
            for (int r = 0; r < 8; ++r) {
                float x = acc[mt * NNT + nt][r] + b;
                if (RELU) x = fmaxf(x, 0.0f);
                dst[(mt * 16 + mb + r) * pitchH + nt * 16 + n0] = (half_t)x;
            }
        }
    }
}

__global__ void __launch_bounds__(128)
mlp_kernel(const float* __restrict__ coor, const float* __restrict__ view,
           const int* __restrict__ ray_id, const half_t* __restrict__ wf,
           const float* bb0, const float* bb1, const float* bb2,
           const float* ba0, const float* ba1, const float* ba2,
           const float* b_lat, const float* b_view, const float* b_c0,
           const float* w_c1, const float* b_c1,
           float* __restrict__ rgb_out) {
    __shared__ WaveLds lds[4];
    const int lane = threadIdx.x & 31;
    const int wv   = threadIdx.x >> 5;
    WaveLds& L = lds[wv];
    const int p0 = (blockIdx.x * 4 + wv) * 32;
    const int p  = p0 + lane;

    // ---- phase 1: coord embedding (all 32 lanes, one point each)
    {
        const float cx = coor[p * 3 + 0], cy = coor[p * 3 + 1], cz = coor[p * 3 + 2];
        float c3[3] = {cx, cy, cz};
        half_t* er = &L.E[lane * 64];
        er[0] = (half_t)cx; er[1] = (half_t)cy; er[2] = (half_t)cz;
#pragma unroll
        for (int fi = 0; fi < 5; ++fi) {
            float fscale = (float)(1 << fi);
#pragma unroll
            for (int j = 0; j < 3; ++j) {
                float a = fscale * c3[j];
                er[3 + 6 * fi + j] = (half_t)__sinf(a);
                er[6 + 6 * fi + j] = (half_t)__cosf(a);
            }
        }
#pragma unroll
        for (int j = 33; j < 64; ++j) er[j] = (half_t)0.0f;
    }
    // ---- phase 2: gathered view embedding (all 32 lanes, one point each)
    {
        const int rid = ray_id[p];
        const float vx = view[rid * 3 + 0], vy = view[rid * 3 + 1], vz = view[rid * 3 + 2];
        float v3[3] = {vx, vy, vz};
        half_t* vr = &L.V[lane * 32];
        vr[0] = (half_t)vx; vr[1] = (half_t)vy; vr[2] = (half_t)vz;
#pragma unroll
        for (int fi = 0; fi < 3; ++fi) {
            float fscale = (float)(1 << fi);
#pragma unroll
            for (int j = 0; j < 3; ++j) {
                float a = fscale * v3[j];
                vr[3 + 6 * fi + j] = (half_t)__sinf(a);
                vr[6 + 6 * fi + j] = (half_t)__cosf(a);
            }
        }
#pragma unroll
        for (int j = 21; j < 32; ++j) vr[j] = (half_t)0.0f;
    }
    __syncthreads();

    const v8f vz8 = {0, 0, 0, 0, 0, 0, 0, 0};
    v8f acc[8];

    // L0: h = relu(emb @ wb0 + bb0)
#pragma unroll
    for (int i = 0; i < 8; ++i) acc[i] = vz8;
    gemm_acc2<2, 4>(acc, L.E, 64, wf + OFF_WB0, lane);
    store_act2<4, true>(acc, L.H1, 64, bb0, lane);
    __syncthreads();
    // L1
#pragma unroll
    for (int i = 0; i < 8; ++i) acc[i] = vz8;
    gemm_acc2<2, 4>(acc, L.H1, 64, wf + OFF_WB1, lane);
    store_act2<4, true>(acc, L.H2, 64, bb1, lane);
    __syncthreads();
    // L2
#pragma unroll
    for (int i = 0; i < 8; ++i) acc[i] = vz8;
    gemm_acc2<2, 4>(acc, L.H2, 64, wf + OFF_WB2, lane);
    store_act2<4, true>(acc, L.H1, 64, bb2, lane);
    __syncthreads();
    // L3: skip concat [emb, h] @ wa0 = emb@wa0[0:33] + h@wa0[33:97]
#pragma unroll
    for (int i = 0; i < 8; ++i) acc[i] = vz8;
    gemm_acc2<2, 4>(acc, L.E, 64, wf + OFF_WA0T, lane);
    gemm_acc2<2, 4>(acc, L.H1, 64, wf + OFF_WA0B, lane);
    store_act2<4, true>(acc, L.H2, 64, ba0, lane);
    __syncthreads();
    // L4
#pragma unroll
    for (int i = 0; i < 8; ++i) acc[i] = vz8;
    gemm_acc2<2, 4>(acc, L.H2, 64, wf + OFF_WA1, lane);
    store_act2<4, true>(acc, L.H1, 64, ba1, lane);
    __syncthreads();
    // L5
#pragma unroll
    for (int i = 0; i < 8; ++i) acc[i] = vz8;
    gemm_acc2<2, 4>(acc, L.H1, 64, wf + OFF_WA2, lane);
    store_act2<4, true>(acc, L.H2, 64, ba2, lane);
    __syncthreads();
    // L6: latent (no relu)
#pragma unroll
    for (int i = 0; i < 8; ++i) acc[i] = vz8;
    gemm_acc2<2, 4>(acc, L.H2, 64, wf + OFF_WLAT, lane);
    store_act2<4, false>(acc, L.H1, 64, b_lat, lane);
    __syncthreads();
    // L7: hv = relu([latent, view_emb] @ w_view + b_view) -> 32 wide (H2 cols 0..31)
#pragma unroll
    for (int i = 0; i < 4; ++i) acc[i] = vz8;
    gemm_acc2<2, 2>(acc, L.H1, 64, wf + OFF_WVT, lane);
    gemm_acc2<1, 2>(acc, L.V, 32, wf + OFF_WVB, lane);
    store_act2<2, true>(acc, L.H2, 64, b_view, lane);
    __syncthreads();
    // L8: hc = relu(hv @ w_c0 + b_c0) -> 16 wide (H1 cols 0..15)
    acc[0] = vz8; acc[1] = vz8;
    gemm_acc2<1, 1>(acc, L.H2, 64, wf + OFF_WC0, lane);
    store_act2<1, true>(acc, L.H1, 64, b_c0, lane);
    __syncthreads();

    // ---- final tiny head 16->3, tanh-squash, write per-point rgb (all lanes)
    {
        const half_t* hr = &L.H1[lane * 64];
        float h16[16];
#pragma unroll
        for (int k = 0; k < 16; ++k) h16[k] = (float)hr[k];
#pragma unroll
        for (int j = 0; j < 3; ++j) {
            float a = b_c1[j];
#pragma unroll
            for (int k = 0; k < 16; ++k) a += h16[k] * w_c1[k * 3 + j];
            rgb_out[p * 3 + j] = 0.5f * (tanhf(a) + 1.0f);
        }
    }
}

// ============================================================== compositing
__global__ void __launch_bounds__(256)
composite_kernel(const int* __restrict__ ray_id, const float* __restrict__ raw_density,
                 const float* __restrict__ rgb, float* __restrict__ out) {
    const int lane = threadIdx.x & 31;
    const int r = blockIdx.x * 8 + (threadIdx.x >> 5);
    if (r >= N_RAYS) return;

    // segment bounds via binary search on sorted ray_id
    int lo = 0, hi = P_TOT;
    while (lo < hi) { int m = (lo + hi) >> 1; if (ray_id[m] < r) lo = m + 1; else hi = m; }
    const int s = lo;
    lo = 0; hi = P_TOT;
    while (lo < hi) { int m = (lo + hi) >> 1; if (ray_id[m] < r + 1) lo = m + 1; else hi = m; }
    const int e = lo;

    float carry = 0.0f, aR = 0.0f, aG = 0.0f, aB = 0.0f;
    for (int base = s; base < e; base += 32) {
        const int idx = base + lane;
        const bool active = idx < e;
        float lt = 0.0f, al = 0.0f;
        if (active) {
            float x = raw_density[idx] + ACT_SHIFT;
            float sp = (x > 20.0f) ? x : log1pf(expf(x));   // softplus
            lt = -sp * INTERVAL;                            // log(1-alpha)
            al = 1.0f - expf(lt);
        }
        // wave32 inclusive scan of lt
        float incl = lt;
#pragma unroll
        for (int d = 1; d < 32; d <<= 1) {
            float t = __shfl_up(incl, d, 32);
            if (lane >= d) incl += t;
        }
        float excl = incl - lt;
        float total = __shfl(incl, 31, 32);
        if (active) {
            float T = expf(carry + excl);    // transmittance
            float w = T * al;
            aR += w * rgb[idx * 3 + 0];
            aG += w * rgb[idx * 3 + 1];
            aB += w * rgb[idx * 3 + 2];
        }
        carry += total;
    }
#pragma unroll
    for (int d = 16; d > 0; d >>= 1) {
        aR += __shfl_down(aR, d, 32);
        aG += __shfl_down(aG, d, 32);
        aB += __shfl_down(aB, d, 32);
    }
    if (lane == 0) {
        float bg = expf(carry);              // alphainv_last (white background)
        out[r * 3 + 0] = aR + bg;
        out[r * 3 + 1] = aG + bg;
        out[r * 3 + 2] = aB + bg;
    }
}

// ============================================================== launch glue
extern "C" void kernel_launch(void* const* d_in, const int* in_sizes, int n_in,
                              void* d_out, int out_size, void* d_ws, size_t ws_size,
                              hipStream_t stream) {
    (void)in_sizes; (void)n_in; (void)out_size; (void)ws_size;
    const float* coor   = (const float*)d_in[0];
    const float* view   = (const float*)d_in[1];
    const float* rawd   = (const float*)d_in[2];
    const int*   rid    = (const int*)d_in[3];
    const float* wb0    = (const float*)d_in[4];
    const float* bb0    = (const float*)d_in[5];
    const float* wb1    = (const float*)d_in[6];
    const float* bb1    = (const float*)d_in[7];
    const float* wb2    = (const float*)d_in[8];
    const float* bb2    = (const float*)d_in[9];
    const float* wa0    = (const float*)d_in[10];
    const float* ba0    = (const float*)d_in[11];
    const float* wa1    = (const float*)d_in[12];
    const float* ba1    = (const float*)d_in[13];
    const float* wa2    = (const float*)d_in[14];
    const float* ba2    = (const float*)d_in[15];
    const float* wlat   = (const float*)d_in[16];
    const float* blat   = (const float*)d_in[17];
    const float* wview  = (const float*)d_in[18];
    const float* bview  = (const float*)d_in[19];
    const float* wc0    = (const float*)d_in[20];
    const float* bc0    = (const float*)d_in[21];
    const float* wc1    = (const float*)d_in[22];
    const float* bc1    = (const float*)d_in[23];

    half_t* wf   = (half_t*)d_ws;
    float* rgbws = (float*)((char*)d_ws + RGB_WS_OFF);
    float* out   = (float*)d_out;

    prep_weights<<<(WF_TOTAL_HALFS + 255) / 256, 256, 0, stream>>>(
        wb0, wb1, wb2, wa0, wa1, wa2, wlat, wview, wc0, wf);

    mlp_kernel<<<P_TOT / 128, 128, 0, stream>>>(
        coor, view, rid, wf,
        bb0, bb1, bb2, ba0, ba1, ba2, blat, bview, bc0, wc1, bc1, rgbws);

    composite_kernel<<<N_RAYS / 8, 256, 0, stream>>>(rid, rawd, rgbws, out);
}
